// VectorQuantizer_30683246362933
// MI455X (gfx1250) — matrix operand order repaired and checked by reading, not compiled, not run on hardware
//
#include <hip/hip_runtime.h>
#include <hip/hip_bf16.h>

// ---------------- problem constants ----------------
constexpr int NT = 65536;   // tokens
constexpr int D  = 512;     // latent dim
constexpr int NC = 4096;    // codebook entries
constexpr int AP = 520;     // padded LDS row stride (ushorts): 1040B -> 4-bank shift/row

// ---------------- vector types ----------------
typedef unsigned int   u32;
typedef unsigned short u16;
typedef unsigned long long u64;
typedef __attribute__((ext_vector_type(4)))  u32    u32x4;
typedef __attribute__((ext_vector_type(2)))  u32    u32x2;
typedef __attribute__((ext_vector_type(4)))  int    i32x4;
typedef __attribute__((ext_vector_type(8)))  int    i32x8;
typedef __attribute__((ext_vector_type(4)))  float  f32x4;
typedef __attribute__((ext_vector_type(8)))  float  v8f;
typedef __attribute__((ext_vector_type(16))) __bf16 v16bf;

struct BF16Frag { u32x4 a, b; };   // 32 bytes == v16bf

#if defined(__has_builtin)
#  if __has_builtin(__builtin_amdgcn_tensor_load_to_lds) && \
      __has_builtin(__builtin_amdgcn_s_wait_tensorcnt)
#    define HAVE_TDM 1
#  endif
#endif
#ifndef HAVE_TDM
#  define HAVE_TDM 0
#endif

static __device__ inline u16 f2bf(float f) {
    u32 u = __builtin_bit_cast(u32, f);
    u32 r = u + 0x7FFFu + ((u >> 16) & 1u);   // round-to-nearest-even
    return (u16)(r >> 16);
}

static __device__ inline v16bf load_frag(const u16* p0, const u16* p1) {
    BF16Frag s;
    s.a = *(const u32x4*)p0;
    s.b = *(const u32x4*)p1;
    return __builtin_bit_cast(v16bf, s);
}

// ---- TDM: DMA one B tile (32 codebook rows x 512 bf16) global -> LDS --------
// D# per CDNA5 ISA ch8: data_size=2B, tile_dim0=512, tile_dim1=32,
// tensor_dim0_stride=512. LDS pad: every 256 DWORDs (one 1024B row) insert
// 4 DWORDs -> effective LDS row stride 1040B == AP ushorts (bank-spread).
static __device__ inline void tdm_load_b_tile(const u16* gsrc, u16* ldst) {
#if HAVE_TDM
    u64 ga = (u64)(uintptr_t)gsrc;
    u32x4 g0;
    g0.x = 1u;                                          // count=1 user D#
    g0.y = (u32)(uintptr_t)ldst;                        // lds_addr (bytes)
    g0.z = (u32)ga;                                     // global_addr[31:0]
    g0.w = ((u32)(ga >> 32) & 0x01FFFFFFu) | (2u << 30);// addr[56:32] | type=2
    i32x8 g1;
    g1[0] = (int)((1u << 16)        // data_size = 2 bytes
                | (1u << 20)        // pad_enable
                | (7u << 22)        // pad_interval: 256 DWORDs (one row)
                | (3u << 25));      // pad_amount:   4 DWORDs (16B)
    g1[1] = (int)(512u << 16);      // tensor_dim0 = 512 (bits 63:48)
    g1[2] = (int)(4096u << 16);     // tensor_dim1 = 4096 (bits 95:80)
    g1[3] = (int)(512u << 16);      // tile_dim0 = 512 (bits 127:112)
    g1[4] = 32;                     // tile_dim1 = 32
    g1[5] = 512;                    // tensor_dim0_stride = 512
    g1[6] = 0;
    g1[7] = 0;
    i32x4 z4 = {0, 0, 0, 0};
#if defined(__clang_major__) && (__clang_major__ >= 23)
    i32x8 z8 = {0, 0, 0, 0, 0, 0, 0, 0};
    __builtin_amdgcn_tensor_load_to_lds(g0, g1, z4, z4, z8, 0);
#else
    __builtin_amdgcn_tensor_load_to_lds(g0, g1, z4, z4, 0);
#endif
#endif
}

// ---------------- kernel 1: normalize codebook twice -> cb (f32), cb2 (bf16) -
__global__ void __launch_bounds__(128)
vq_norm_cb(const float* __restrict__ cbp, float* __restrict__ cbf,
           u16* __restrict__ cbbf) {
    const int row = blockIdx.x, t = threadIdx.x;
    f32x4 v = ((const f32x4*)(cbp + (size_t)row * D))[t];
    float p = v.x * v.x + v.y * v.y + v.z * v.z + v.w * v.w;
    #pragma unroll
    for (int m = 16; m >= 1; m >>= 1) p += __shfl_xor(p, m, 32);
    __shared__ float sh[4];
    if ((t & 31) == 0) sh[t >> 5] = p;
    __syncthreads();
    float s    = sh[0] + sh[1] + sh[2] + sh[3];
    float inv1 = 1.0f / (sqrtf(s) + 1e-8f);
    f32x4 cb   = v * inv1;                        // normalize(param)
    float s2   = s * inv1 * inv1;
    float inv2 = 1.0f / (sqrtf(s2) + 1e-8f);
    f32x4 cb2  = cb * inv2;                       // normalize(normalize(param))
    ((f32x4*)(cbf + (size_t)row * D))[t] = cb;    // gather source (z uses cb)
    u32x2 pk;
    pk.x = (u32)f2bf(cb2.x) | ((u32)f2bf(cb2.y) << 16);
    pk.y = (u32)f2bf(cb2.z) | ((u32)f2bf(cb2.w) << 16);
    ((u32x2*)(cbbf + (size_t)row * D))[t] = pk;   // GEMM B operand
}

// ---------------- kernel 2: normalize x -> xn (f32 into d_out) + bf16 copy ---
__global__ void __launch_bounds__(128)
vq_norm_x(const float* __restrict__ x, float* __restrict__ xn_out,
          u16* __restrict__ xnbf) {
    const int row = blockIdx.x, t = threadIdx.x;
    f32x4 v = ((const f32x4*)(x + (size_t)row * D))[t];
    float p = v.x * v.x + v.y * v.y + v.z * v.z + v.w * v.w;
    #pragma unroll
    for (int m = 16; m >= 1; m >>= 1) p += __shfl_xor(p, m, 32);
    __shared__ float sh[4];
    if ((t & 31) == 0) sh[t >> 5] = p;
    __syncthreads();
    float s = sh[0] + sh[1] + sh[2] + sh[3];
    float inv = 1.0f / (sqrtf(s) + 1e-8f);
    f32x4 o = v * inv;
    ((f32x4*)(xn_out + (size_t)row * D))[t] = o;
    u32x2 pk;
    pk.x = (u32)f2bf(o.x) | ((u32)f2bf(o.y) << 16);
    pk.y = (u32)f2bf(o.z) | ((u32)f2bf(o.w) << 16);
    ((u32x2*)(xnbf + (size_t)row * D))[t] = pk;
}

// ---- per-wave compute: 16 rows x 32 codes, shared A fragment, 2 WMMA chains -
static __device__ inline void compute_tile(const u16* __restrict__ arow,
                                           const u16* __restrict__ bbase,
                                           int half, int l16, int c0,
                                           float* bestV, int* bestI) {
    const u16* b0row = bbase + l16 * AP;          // code c0 + l16
    const u16* b1row = bbase + (16 + l16) * AP;   // code c0 + 16 + l16
    v8f acc0 = {}, acc1 = {};
    #pragma unroll
    for (int k0 = 0; k0 < D; k0 += 32) {
        v16bf a  = load_frag(arow + k0 + half * 16,
                             arow + k0 + half * 16 + 8);
        v16bf b0 = load_frag(b0row + k0 + half * 16,
                             b0row + k0 + half * 16 + 8);
        v16bf b1 = load_frag(b1row + k0 + half * 16,
                             b1row + k0 + half * 16 + 8);
        acc0 = __builtin_amdgcn_wmma_f32_16x16x32_bf16(
            false, a, false, b0, (short)0, acc0, false, false);
        acc1 = __builtin_amdgcn_wmma_f32_16x16x32_bf16(
            false, a, false, b1, (short)0, acc1, false, false);
    }
    #pragma unroll
    for (int v = 0; v < 8; ++v) {   // lower code first => argmin first-hit ties
        float s0 = acc0[v];
        if (s0 > bestV[v]) { bestV[v] = s0; bestI[v] = c0 + l16; }
        float s1 = acc1[v];
        if (s1 > bestV[v]) { bestV[v] = s1; bestI[v] = c0 + 16 + l16; }
    }
}

// ---------------- kernel 3: fused bf16 WMMA GEMM + running argmax ------------
// block = 256 threads (8 waves) = 128 token rows. A tile in LDS once; B tiles
// (32 codes) DMA'd by the Tensor Data Mover, double-buffered.
__global__ void __launch_bounds__(256)
vq_gemm_argmax(const u16* __restrict__ xnbf, const u16* __restrict__ cbbf,
               int* __restrict__ idxout) {
    extern __shared__ u16 smem[];
    u16* Asm_ = smem;                 // [128][AP]
    u16* Bsm_ = smem + 128 * AP;      // [2][32][AP]

    const int tid  = threadIdx.x;
    const int w    = tid >> 5;
    const int lane = tid & 31;
    const int half = lane >> 4;
    const int l16  = lane & 15;
    const int m0   = blockIdx.x * 128;

#if HAVE_TDM
    if (w == 0) tdm_load_b_tile(cbbf, Bsm_);            // prefetch tile 0
#endif
    // cooperative A-tile load: 128 rows x 512 bf16 (overlaps with TDM)
    for (int i = tid; i < 128 * 64; i += 256) {
        int r = i >> 6, j = i & 63;
        ((u32x4*)(Asm_ + r * AP))[j] =
            ((const u32x4*)(xnbf + (size_t)(m0 + r) * D))[j];
    }

    const u16* arow = Asm_ + (w * 16 + l16) * AP;

    float bestV[8];
    int   bestI[8];
    #pragma unroll
    for (int v = 0; v < 8; ++v) { bestV[v] = -3.4e38f; bestI[v] = 0; }

#if HAVE_TDM
    if (w == 0) __builtin_amdgcn_s_wait_tensorcnt(0);   // tile 0 resident
    __syncthreads();
    for (int c0 = 0; c0 < NC; c0 += 32) {
        const int par = (c0 >> 5) & 1;
        if (w == 0 && (c0 + 32) < NC)                   // prefetch tile t+1
            tdm_load_b_tile(cbbf + (size_t)(c0 + 32) * D,
                            Bsm_ + (par ^ 1) * (32 * AP));
        compute_tile(arow, Bsm_ + par * (32 * AP), half, l16, c0, bestV, bestI);
        if (w == 0) __builtin_amdgcn_s_wait_tensorcnt(0);
        __syncthreads();
    }
#else
    for (int c0 = 0; c0 < NC; c0 += 32) {
        __syncthreads();
        for (int i = tid; i < 32 * 64; i += 256) {      // cooperative B copy
            int r = i >> 6, j = i & 63;
            ((u32x4*)(Bsm_ + r * AP))[j] =
                ((const u32x4*)(cbbf + (size_t)(c0 + r) * D))[j];
        }
        __syncthreads();
        compute_tile(arow, Bsm_, half, l16, c0, bestV, bestI);
    }
#endif

    // reduce across the 16 lanes of each half; tie -> smaller index (argmin rule)
    #pragma unroll
    for (int m = 1; m <= 8; m <<= 1) {
        #pragma unroll
        for (int v = 0; v < 8; ++v) {
            float ov = __shfl_xor(bestV[v], m, 32);
            int   oi = __shfl_xor(bestI[v], m, 32);
            if (ov > bestV[v] || (ov == bestV[v] && oi < bestI[v])) {
                bestV[v] = ov; bestI[v] = oi;
            }
        }
    }
    if (l16 == 0) {
        int rowBase = m0 + w * 16 + half * 8;
        #pragma unroll
        for (int v = 0; v < 8; ++v) idxout[rowBase + v] = bestI[v];
    }
}

// ---------------- kernel 4: gather z = cb[idx]; z_q = xn + (z - xn) ----------
__global__ void __launch_bounds__(128)
vq_gather(const int* __restrict__ idx, const float* __restrict__ cbf,
          const float* __restrict__ xn, float* __restrict__ zq,
          float* __restrict__ z, float* __restrict__ idxf) {
    const int row = blockIdx.x, t = threadIdx.x;
    const int k = idx[row];
    f32x4 zv = ((const f32x4*)(cbf + (size_t)k * D))[t];
    f32x4 xv = ((const f32x4*)(xn + (size_t)row * D))[t];
    f32x4 q  = xv + (zv - xv);
    ((f32x4*)(z  + (size_t)row * D))[t] = zv;
    ((f32x4*)(zq + (size_t)row * D))[t] = q;
    if (t == 0) idxf[row] = (float)k;
}

// ---------------- launcher ---------------------------------------------------
extern "C" void kernel_launch(void* const* d_in, const int* in_sizes, int n_in,
                              void* d_out, int out_size, void* d_ws, size_t ws_size,
                              hipStream_t stream) {
    (void)in_sizes; (void)n_in; (void)out_size; (void)ws_size;
    const float* x   = (const float*)d_in[0];
    const float* cbp = (const float*)d_in[1];
    // d_in[2] = training flag; reference path uses training=0 (dropout identity)

    float* out  = (float*)d_out;
    float* zq   = out;
    float* z    = out + (size_t)NT * D;
    float* xn   = out + 2 * (size_t)NT * D;
    float* idxf = out + 3 * (size_t)NT * D;

    char* ws = (char*)d_ws;
    u16*  xnbf = (u16*)ws;                                        // NT*D bf16
    u16*  cbbf = (u16*)(ws + (size_t)NT * D * 2);                 // NC*D bf16
    float* cbf = (float*)(ws + (size_t)NT * D * 2
                             + (size_t)NC * D * 2);               // NC*D f32
    int*  idx  = (int*)(ws + (size_t)NT * D * 2
                           + (size_t)NC * D * 2
                           + (size_t)NC * D * 4);                 // NT i32

    vq_norm_cb<<<NC, 128, 0, stream>>>(cbp, cbf, cbbf);
    vq_norm_x <<<NT, 128, 0, stream>>>(x, xn, xnbf);

    constexpr size_t smem = (size_t)(128 * AP + 2 * 32 * AP) * sizeof(u16); // 199680 B
    vq_gemm_argmax<<<NT / 128, 256, smem, stream>>>(xnbf, cbbf, idx);

    vq_gather<<<NT, 128, 0, stream>>>(idx, cbf, xn, zq, z, idxf);
}
